// CrossAttentionBlock_28235115003907
// MI455X (gfx1250) — matrix operand order repaired and checked by reading, not compiled, not run on hardware
//
#include <hip/hip_runtime.h>
#include <math.h>

// Problem constants (match reference)
#define BB 4
#define LL 2048
#define DD 1024
#define HH 16
#define HD 64
#define MROWS (BB * LL)   // 8192

typedef _Float16 v8h  __attribute__((ext_vector_type(8)));
typedef _Float16 v16h __attribute__((ext_vector_type(16)));
typedef float    v8f  __attribute__((ext_vector_type(8)));

static __device__ __forceinline__ v16h cat8(v8h lo, v8h hi) {
  return __builtin_shufflevector(lo, hi, 0,1,2,3,4,5,6,7,8,9,10,11,12,13,14,15);
}

// 16-byte async copy global -> LDS (CDNA5 async engine, tracked by ASYNCcnt).
// LDS operand: 32-bit LDS byte address (low 32 bits of the generic pointer).
static __device__ __forceinline__ void async_cp16(void* lds, const void* gptr) {
  unsigned loff = (unsigned)(unsigned long long)lds;
  asm volatile("global_load_async_to_lds_b128 %0, %1, off"
               :: "v"(loff), "v"((unsigned long long)gptr) : "memory");
}
static __device__ __forceinline__ void wait_async0() {
  asm volatile("s_wait_asynccnt 0x0" ::: "memory");
}

// ---------------------------------------------------------------------------
// Transposing f32 -> f16 weight conversion: W[K,N] (f32) -> WT[N,K] (f16).
// Tile 32x32 through LDS so both sides are coalesced. Grid (N/32, K/32).
// ---------------------------------------------------------------------------
__global__ __launch_bounds__(256)
void transpose_cvt_f16(const float* __restrict__ W, _Float16* __restrict__ WT,
                       int K, int N) {
  __shared__ float t[32][33];
  const int kb = blockIdx.y * 32, nb = blockIdx.x * 32;
  const int c = threadIdx.x & 31, r0 = threadIdx.x >> 5;   // 8 rows/pass
#pragma unroll
  for (int p = 0; p < 4; ++p)
    t[r0 + p * 8][c] = W[(size_t)(kb + r0 + p * 8) * N + nb + c];
  __syncthreads();
#pragma unroll
  for (int p = 0; p < 4; ++p)
    WT[(size_t)(nb + r0 + p * 8) * K + kb + c] = (_Float16)t[c][r0 + p * 8];
}

// ---------------------------------------------------------------------------
// LayerNorm: f32 [rows, 1024] -> f16 [rows, 1024]
// ---------------------------------------------------------------------------
__global__ __launch_bounds__(256)
void layernorm_f16(const float* __restrict__ x, const float* __restrict__ g,
                   const float* __restrict__ be, _Float16* __restrict__ out) {
  const int row = blockIdx.x;
  const float* xr = x + (size_t)row * DD;
  float vals[4];
  float s = 0.f, sq = 0.f;
#pragma unroll
  for (int i = 0; i < 4; ++i) {
    float v = xr[threadIdx.x + i * 256];
    vals[i] = v; s += v; sq += v * v;
  }
#pragma unroll
  for (int off = 16; off >= 1; off >>= 1) {
    s  += __shfl_xor(s,  off, 32);
    sq += __shfl_xor(sq, off, 32);
  }
  __shared__ float ss[8], ssq[8];
  const int wave = threadIdx.x >> 5, lane = threadIdx.x & 31;
  if (lane == 0) { ss[wave] = s; ssq[wave] = sq; }
  __syncthreads();
  s = 0.f; sq = 0.f;
#pragma unroll
  for (int i = 0; i < 8; ++i) { s += ss[i]; sq += ssq[i]; }
  const float mean = s * (1.0f / DD);
  const float var  = sq * (1.0f / DD) - mean * mean;
  const float rstd = rsqrtf(var + 1e-5f);
#pragma unroll
  for (int i = 0; i < 4; ++i) {
    int c = threadIdx.x + i * 256;
    out[(size_t)row * DD + c] = (_Float16)((vals[i] - mean) * rstd * g[c] + be[c]);
  }
}

// ---------------------------------------------------------------------------
// Tiled WMMA GEMM: out[M,N] = A[M,K](f16) @ WT[N,K](f16, pre-transposed) + bias
// Block tile 128x128, 8 waves, each wave 32x64 (2x4 of 16x16 C-tiles).
// Double-buffered LDS fed by async global->LDS copies (ASYNCcnt).
// flags bit0: store f16, bit1: exact GELU. residual (f32, [M,N]) optional.
// ---------------------------------------------------------------------------
__global__ __launch_bounds__(256)
void gemm_f16_wmma(const _Float16* __restrict__ A, const _Float16* __restrict__ WT,
                   const float* __restrict__ bias, const float* __restrict__ residual,
                   void* __restrict__ out, int M, int N, int K, int flags) {
  __shared__ __align__(16) _Float16 Alds[2][128][48];   // [buf][row][k]
  __shared__ __align__(16) _Float16 Wlds[2][128][48];   // [buf][n][k]

  const int tid = threadIdx.x;
  const int lane = tid & 31;
  const int wave = tid >> 5;
  const int laneHalf = lane >> 4;   // 0/1
  const int laneCol  = lane & 15;
  const int waveM = wave & 3;       // 4 M-groups of 32 rows
  const int waveN = wave >> 2;      // 2 N-groups of 64 cols
  const int mBase = blockIdx.y * 128;
  const int nBase = blockIdx.x * 128;

  const int sr  = tid >> 2;          // 0..63
  const int sch = (tid & 3) * 8;     // 0,8,16,24

  auto stage = [&](int buf, int k0) {
#pragma unroll
    for (int p = 0; p < 2; ++p) {
      const int row = p * 64 + sr;
      async_cp16(&Alds[buf][row][sch], A  + (size_t)(mBase + row) * K + k0 + sch);
      async_cp16(&Wlds[buf][row][sch], WT + (size_t)(nBase + row) * K + k0 + sch);
    }
  };

  v8f acc[2][4] = {};

  stage(0, 0);
  wait_async0();
  __syncthreads();

  int cur = 0;
  for (int k0 = 0; k0 < K; k0 += 32) {
    if (k0 + 32 < K) stage(cur ^ 1, k0 + 32);   // prefetch next tile (async)

    // --- A fragments (16x32): lane row = laneCol, halves split by laneHalf ---
    v16h af[2];
#pragma unroll
    for (int mi = 0; mi < 2; ++mi) {
      const int row = waveM * 32 + mi * 16 + laneCol;
      v8h lo = *(const v8h*)(&Alds[cur][row][laneHalf * 8]);
      v8h hi = *(const v8h*)(&Alds[cur][row][16 + laneHalf * 8]);
      af[mi] = cat8(lo, hi);
    }
    // --- B fragments (32x16): lane col = laneCol, k-range split by laneHalf ---
#pragma unroll
    for (int ni = 0; ni < 4; ++ni) {
      const int col = waveN * 64 + ni * 16 + laneCol;
      v8h lo = *(const v8h*)(&Wlds[cur][col][laneHalf * 8]);
      v8h hi = *(const v8h*)(&Wlds[cur][col][16 + laneHalf * 8]);
      v16h bf = cat8(lo, hi);
#pragma unroll
      for (int mi = 0; mi < 2; ++mi)
        acc[mi][ni] = __builtin_amdgcn_wmma_f32_16x16x32_f16(
            false, af[mi], false, bf, (short)0, acc[mi][ni], false, false);
    }

    wait_async0();       // my async copies done; barrier makes all waves' visible
    __syncthreads();
    cur ^= 1;
  }

  // --- epilogue: bias (+gelu) (+residual), store f16/f32 ---
#pragma unroll
  for (int mi = 0; mi < 2; ++mi) {
#pragma unroll
    for (int ni = 0; ni < 4; ++ni) {
      const int gCol = nBase + waveN * 64 + ni * 16 + laneCol;
      const float bv = bias[gCol];
#pragma unroll
      for (int r = 0; r < 8; ++r) {
        const int gRow = mBase + waveM * 32 + mi * 16 + laneHalf * 8 + r;
        float v = acc[mi][ni][r] + bv;
        if (flags & 2) v = 0.5f * v * (1.0f + erff(v * 0.70710678118654752f));
        if (residual) v += residual[(size_t)gRow * N + gCol];
        if (flags & 1) ((_Float16*)out)[(size_t)gRow * N + gCol] = (_Float16)v;
        else           ((float*)out)[(size_t)gRow * N + gCol] = v;
      }
    }
  }
}

// ---------------------------------------------------------------------------
// RoPE in-place on f16 buffer laid out [B, L, rowStride] with heads at h*64.
// One thread per (b,l,h,d<32) pair.
// ---------------------------------------------------------------------------
__global__ void rope_inplace(_Float16* __restrict__ q, const float* __restrict__ cs,
                             const float* __restrict__ sn, int rowStride) {
  size_t idx = (size_t)blockIdx.x * 256 + threadIdx.x;
  const int d = idx & 31;            idx >>= 5;
  const int h = idx % HH;            idx /= HH;
  const int l = idx % LL;
  const size_t b = idx / LL;
  const size_t base = ((size_t)(b * LL + l)) * rowStride + h * HD;
  const float x1 = (float)q[base + d];
  const float x2 = (float)q[base + d + 32];
  const float c1 = cs[l * HD + d],      c2 = cs[l * HD + d + 32];
  const float s1 = sn[l * HD + d],      s2 = sn[l * HD + d + 32];
  q[base + d]      = (_Float16)(x1 * c1 - x2 * s1);
  q[base + d + 32] = (_Float16)(x2 * c2 + x1 * s2);
}

// ---------------------------------------------------------------------------
// Flash attention. Grid: (L/128, H, B). Block: 256 threads = 8 waves.
// Each wave owns 16 queries; streams KV in 32-key chunks with online softmax.
// ---------------------------------------------------------------------------
__global__ __launch_bounds__(256)
void attention_fwd(const _Float16* __restrict__ q, const _Float16* __restrict__ kv,
                   _Float16* __restrict__ out) {
  __shared__ __align__(16) _Float16 Klds[32][72];      // [key][hd]
  __shared__ __align__(16) _Float16 Vlds[64][48];      // [hd][key] transposed
  __shared__ __align__(16) _Float16 Plds[8][16][48];   // per-wave P scratch

  const int qt = blockIdx.x, h = blockIdx.y, b = blockIdx.z;
  const int tid = threadIdx.x, lane = tid & 31, wave = tid >> 5;
  const int laneHalf = lane >> 4, laneCol = lane & 15;
  const size_t kvStride = 2 * DD;
  const float SCALE = 0.125f;   // 1/sqrt(64)

  // Q fragments (16 rows x 64 hd, as two 16x32 A-fragments), kept in VGPRs
  const int qRow = qt * 128 + wave * 16 + laneCol;
  const size_t qBase = ((size_t)(b * LL + qRow)) * DD + h * HD;
  v16h qa[2];
#pragma unroll
  for (int c = 0; c < 2; ++c) {
    v8h lo = *(const v8h*)(q + qBase + c * 32 + laneHalf * 8);
    v8h hi = *(const v8h*)(q + qBase + c * 32 + 16 + laneHalf * 8);
    qa[c] = cat8(lo, hi);
  }

  v8f o[4] = {};
  float mrun[8], lrun[8];
#pragma unroll
  for (int r = 0; r < 8; ++r) { mrun[r] = -1e30f; lrun[r] = 0.f; }

  for (int kv0 = 0; kv0 < LL; kv0 += 32) {
    __syncthreads();
    {   // stage K (row-major) and V (transposed)
      const int row = tid >> 3;          // 0..31
      const int ch  = (tid & 7) * 8;     // 0..56
      const size_t rbase = ((size_t)(b * LL + kv0 + row)) * kvStride + h * HD;
      v8h k8 = *(const v8h*)(kv + rbase + ch);
      *(v8h*)(&Klds[row][ch]) = k8;
      v8h v8v = *(const v8h*)(kv + rbase + DD + ch);
#pragma unroll
      for (int j = 0; j < 8; ++j) Vlds[ch + j][row] = v8v[j];
    }
    __syncthreads();

    // S = Q @ K^T for 2 key tiles of 16 (k-dim 64 = two 32-chunks)
    v8f s[2];
#pragma unroll
    for (int kt = 0; kt < 2; ++kt) {
      const int key = kt * 16 + laneCol;
      v8f z = {};
#pragma unroll
      for (int c = 0; c < 2; ++c) {
        v8h lo = *(const v8h*)(&Klds[key][c * 32 + laneHalf * 8]);
        v8h hi = *(const v8h*)(&Klds[key][c * 32 + 16 + laneHalf * 8]);
        v16h kb = cat8(lo, hi);
        z = __builtin_amdgcn_wmma_f32_16x16x32_f16(false, qa[c], false, kb,
                                                   (short)0, z, false, false);
      }
      s[kt] = z;
    }

    // online softmax update (per C-row r; lane halves carry rows r / r+8)
#pragma unroll
    for (int r = 0; r < 8; ++r) {
      float a0 = s[0][r] * SCALE, a1 = s[1][r] * SCALE;
      float m = fmaxf(a0, a1);
#pragma unroll
      for (int off = 1; off < 16; off <<= 1) m = fmaxf(m, __shfl_xor(m, off, 32));
      const float mnew = fmaxf(mrun[r], m);
      const float alpha = __expf(mrun[r] - mnew);
      const float p0 = __expf(a0 - mnew);
      const float p1 = __expf(a1 - mnew);
      s[0][r] = p0; s[1][r] = p1;
      float rs = p0 + p1;
#pragma unroll
      for (int off = 1; off < 16; off <<= 1) rs += __shfl_xor(rs, off, 32);
      lrun[r] = lrun[r] * alpha + rs;
      mrun[r] = mnew;
#pragma unroll
      for (int t = 0; t < 4; ++t) o[t][r] *= alpha;
    }

    // P (C layout) -> per-wave LDS -> A-fragment layout (wave-internal, in-order)
#pragma unroll
    for (int r = 0; r < 8; ++r) {
      const int prow = laneHalf * 8 + r;
      Plds[wave][prow][laneCol]      = (_Float16)s[0][r];
      Plds[wave][prow][16 + laneCol] = (_Float16)s[1][r];
    }
    asm volatile("s_wait_dscnt 0x0" ::: "memory");
    {
      v8h plo = *(const v8h*)(&Plds[wave][laneCol][laneHalf * 8]);
      v8h phi = *(const v8h*)(&Plds[wave][laneCol][16 + laneHalf * 8]);
      v16h pa = cat8(plo, phi);
#pragma unroll
      for (int t = 0; t < 4; ++t) {
        const int col = t * 16 + laneCol;
        v8h vlo = *(const v8h*)(&Vlds[col][laneHalf * 8]);
        v8h vhi = *(const v8h*)(&Vlds[col][16 + laneHalf * 8]);
        v16h vb = cat8(vlo, vhi);
        o[t] = __builtin_amdgcn_wmma_f32_16x16x32_f16(false, pa, false, vb,
                                                      (short)0, o[t], false, false);
      }
    }
  }

  // normalize and store [B,L,H*HD] f16
#pragma unroll
  for (int t = 0; t < 4; ++t) {
#pragma unroll
    for (int r = 0; r < 8; ++r) {
      const int rowq = qt * 128 + wave * 16 + laneHalf * 8 + r;
      const float v = o[t][r] / lrun[r];
      out[((size_t)(b * LL + rowq)) * DD + h * HD + t * 16 + laneCol] = (_Float16)v;
    }
  }
}

// ---------------------------------------------------------------------------
// Host-side orchestration
// ---------------------------------------------------------------------------
extern "C" void kernel_launch(void* const* d_in, const int* in_sizes, int n_in,
                              void* d_out, int out_size, void* d_ws, size_t ws_size,
                              hipStream_t stream) {
  (void)in_sizes; (void)n_in; (void)out_size; (void)ws_size;
  const float* x     = (const float*)d_in[0];
  const float* ctx   = (const float*)d_in[1];
  const float* rc    = (const float*)d_in[2];
  const float* rs    = (const float*)d_in[3];
  const float* Wq    = (const float*)d_in[4];
  const float* bq    = (const float*)d_in[5];
  const float* Wkv   = (const float*)d_in[6];
  const float* bkv   = (const float*)d_in[7];
  const float* Wo    = (const float*)d_in[8];
  const float* bo    = (const float*)d_in[9];
  const float* g_q   = (const float*)d_in[10];
  const float* be_q  = (const float*)d_in[11];
  const float* g_kv  = (const float*)d_in[12];
  const float* be_kv = (const float*)d_in[13];
  const float* g_ffn = (const float*)d_in[14];
  const float* be_ffn= (const float*)d_in[15];
  const float* W1    = (const float*)d_in[16];
  const float* b1    = (const float*)d_in[17];
  const float* W2    = (const float*)d_in[18];
  const float* b2    = (const float*)d_in[19];
  float* out = (float*)d_out;

  char* w = (char*)d_ws;
  auto alloc = [&](size_t bytes) {
    char* p = w; w += (bytes + 255) & ~(size_t)255; return p;
  };
  _Float16* wqT  = (_Float16*)alloc((size_t)DD * DD * 2);        // [N=D, K=D]
  _Float16* wkvT = (_Float16*)alloc((size_t)2 * DD * DD * 2);    // [N=2D, K=D]
  _Float16* woT  = (_Float16*)alloc((size_t)DD * DD * 2);        // [N=D, K=D]
  _Float16* w1T  = (_Float16*)alloc((size_t)4 * DD * DD * 2);    // [N=4D, K=D]
  _Float16* w2T  = (_Float16*)alloc((size_t)DD * 4 * DD * 2);    // [N=D, K=4D]
  _Float16* xn16 = (_Float16*)alloc((size_t)MROWS * DD * 2);     // also attn buf
  _Float16* cn16 = (_Float16*)alloc((size_t)MROWS * DD * 2);     // also ffn-norm buf
  _Float16* q16  = (_Float16*)alloc((size_t)MROWS * DD * 2);
  _Float16* kv16 = (_Float16*)alloc((size_t)MROWS * 2 * DD * 2);
  float*    x2   = (float*)   alloc((size_t)MROWS * DD * 4);
  _Float16* h116 = (_Float16*)alloc((size_t)MROWS * 4 * DD * 2);
  _Float16* attn16 = xn16;   // dead after Q-GEMM
  _Float16* hn16   = cn16;   // dead after KV-GEMM

  // 1) weights -> f16, transposed to [N,K] so GEMM staging is copy-only
  transpose_cvt_f16<<<dim3(DD/32,   DD/32),   256, 0, stream>>>(Wq,  wqT,  DD,   DD);
  transpose_cvt_f16<<<dim3(2*DD/32, DD/32),   256, 0, stream>>>(Wkv, wkvT, DD,   2*DD);
  transpose_cvt_f16<<<dim3(DD/32,   DD/32),   256, 0, stream>>>(Wo,  woT,  DD,   DD);
  transpose_cvt_f16<<<dim3(4*DD/32, DD/32),   256, 0, stream>>>(W1,  w1T,  DD,   4*DD);
  transpose_cvt_f16<<<dim3(DD/32,   4*DD/32), 256, 0, stream>>>(W2,  w2T,  4*DD, DD);

  // 2) layernorms -> f16
  layernorm_f16<<<MROWS, 256, 0, stream>>>(x,   g_q,  be_q,  xn16);
  layernorm_f16<<<MROWS, 256, 0, stream>>>(ctx, g_kv, be_kv, cn16);

  // 3) Q / KV projections (f16 out)
  gemm_f16_wmma<<<dim3(DD/128, MROWS/128), 256, 0, stream>>>(
      xn16, wqT, bq, nullptr, q16, MROWS, DD, DD, /*f16*/1);
  gemm_f16_wmma<<<dim3(2*DD/128, MROWS/128), 256, 0, stream>>>(
      cn16, wkvT, bkv, nullptr, kv16, MROWS, 2*DD, DD, 1);

  // 4) RoPE in place on Q and the K half of KV
  const int ropeBlocks = (BB * LL * HH * 32) / 256;
  rope_inplace<<<ropeBlocks, 256, 0, stream>>>(q16, rc, rs, DD);
  rope_inplace<<<ropeBlocks, 256, 0, stream>>>(kv16, rc, rs, 2*DD);

  // 5) flash attention
  attention_fwd<<<dim3(LL/128, HH, BB), 256, 0, stream>>>(q16, kv16, attn16);

  // 6) output projection + residual(x) -> x2 (f32)
  gemm_f16_wmma<<<dim3(DD/128, MROWS/128), 256, 0, stream>>>(
      attn16, woT, bo, x, x2, MROWS, DD, DD, /*f32 out*/0);

  // 7) FFN
  layernorm_f16<<<MROWS, 256, 0, stream>>>(x2, g_ffn, be_ffn, hn16);
  gemm_f16_wmma<<<dim3(4*DD/128, MROWS/128), 256, 0, stream>>>(
      hn16, w1T, b1, nullptr, h116, MROWS, 4*DD, DD, /*f16 + gelu*/1 | 2);
  gemm_f16_wmma<<<dim3(DD/128, MROWS/128), 256, 0, stream>>>(
      h116, w2T, b2, x2, out, MROWS, DD, 4*DD, /*f32 out + residual*/0);
}
